// GraphTransformerWithYAtomMapPosEmbInefficient_9612136808542
// MI455X (gfx1250) — matrix operand order repaired and checked by reading, not compiled, not run on hardware
//
#include <hip/hip_runtime.h>
#include <hip/hip_bf16.h>

typedef __attribute__((ext_vector_type(16))) _Float16 v16h;
typedef __attribute__((ext_vector_type(8)))  _Float16 v8h;
typedef __attribute__((ext_vector_type(8)))  float    v8f;

union V16U { v16h v; v8h h[2]; };

#define NB   64
#define NN   128
#define DEIN 5
#define DXIN 16
#define HX   256
#define DXH  256
#define HE   128
#define EOUT 64
#define HY   128
#define YOUT 64

__device__ __forceinline__ v8f wmma16x16x32(v16h a, v16h b, v8f c) {
  // 8 args: (neg_a, A, neg_b, B, c_mod, C, reuse_a, reuse_b)
  return __builtin_amdgcn_wmma_f32_16x16x32_f16(false, a, false, b, (short)0, c, false, false);
}

// ---------------------------------------------------------------- map kernel
__global__ __launch_bounds__(128) void map_kernel(const int* __restrict__ am,
                                                  const int* __restrict__ mol,
                                                  int* __restrict__ mp) {
  int b = blockIdx.x;
  int j = threadIdx.x;
  const int* molr = mol + b * NN;
  const int* amr  = am  + b * NN;
  int pa = 0;
  for (int i = 0; i < NN; ++i) pa = max(pa, molr[i]);
  int a = amr[j];
  int m = -1;
  if (molr[j] != pa && a > 0) {
    for (int i = 0; i < NN; ++i) {
      if (molr[i] == pa && amr[i] == a) { m = i; break; }
    }
  }
  mp[b * NN + j] = m;
}

// ---------------------------------------------------------------- Eh kernel
// block = (kt, jt, b); 256 threads = 8 waves; wave handles 2 M-tiles (j rows)
// Weights staged in fragment-major f16 layout so every B-frag is one aligned
// 32B LDS load; hidden tile rows are 8-half aligned so A-frags are 2x b128.
#define EHS 136   // hidden row stride in halves (68 dwords -> bank-conflict free)
__global__ __launch_bounds__(256) void eh_kernel(const float* __restrict__ E,
                                                 const float* __restrict__ We1,
                                                 const float* __restrict__ be1,
                                                 const float* __restrict__ We2,
                                                 const float* __restrict__ be2,
                                                 const int*   __restrict__ mp,
                                                 float* __restrict__ EhOut) {
  __shared__ __attribute__((aligned(32))) _Float16 sW1f[8 * 32 * 16];       // [t][lane][i]
  __shared__ __attribute__((aligned(32))) _Float16 sW2f[4 * 4 * 32 * 16];   // [ks][nt][lane][i]
  __shared__ __attribute__((aligned(32))) _Float16 sH[8][16 * EHS];
  __shared__ float sb1[HE];
  __shared__ float sb2[EOUT];

  int tid = threadIdx.x;
  // stage We1 (K padded 10->32) in fragment-major layout
  for (int idx = tid; idx < 8 * 32 * 16; idx += 256) {
    int i = idx & 15, lane_s = (idx >> 4) & 31, t = idx >> 9;
    int k = (i & 7) + 16 * (i >> 3) + 8 * (lane_s >> 4);
    int col = t * 16 + (lane_s & 15);
    sW1f[idx] = (k < 10) ? (_Float16)We1[k * HE + col] : (_Float16)0.f;
  }
  // stage We2 in fragment-major layout
  for (int idx = tid; idx < 4 * 4 * 32 * 16; idx += 256) {
    int i = idx & 15, lane_s = (idx >> 4) & 31, nt = (idx >> 9) & 3, ks = idx >> 11;
    int k = ks * 32 + (i & 7) + 16 * (i >> 3) + 8 * (lane_s >> 4);
    int col = nt * 16 + (lane_s & 15);
    sW2f[idx] = (_Float16)We2[k * EOUT + col];
  }
  if (tid < HE)   sb1[tid] = be1[tid];
  if (tid < EOUT) sb2[tid] = be2[tid];
  __syncthreads();

  int b  = blockIdx.z, jt = blockIdx.y, kt = blockIdx.x;
  int wave = tid >> 5, lane = tid & 31;
  int laneHi = lane >> 4, ln = lane & 15;
  _Float16* hbuf = sH[wave];

  for (int sub = 0; sub < 2; ++sub) {
    int mt = wave * 2 + sub;          // j_local for this M-tile
    int j  = jt * 16 + mt;
    int mj = mp[b * NN + j];
    int krow = kt * 16 + ln;          // this lane's A-row (k index)
    int mk = mp[b * NN + krow];
    bool pr = (mj >= 0) && (mk >= 0);

    float ec[10];
    const float* ep = E + (((size_t)(b * NN + j)) * NN + krow) * DEIN;
#pragma unroll
    for (int c = 0; c < 5; ++c) ec[c] = ep[c];
    if (pr) {
      const float* ep2 = E + (((size_t)(b * NN + mj)) * NN + mk) * DEIN;
#pragma unroll
      for (int c = 0; c < 5; ++c) ec[5 + c] = ep2[c];
    } else {
      ec[5] = 1.f;
#pragma unroll
      for (int c = 6; c < 10; ++c) ec[c] = 0.f;
    }

    // ---- layer 1: Ec(16x10 pad 32) @ We1(32x128), single K-step
    v16h a;
#pragma unroll
    for (int i = 0; i < 16; ++i) {
      int k = (i & 7) + ((i >> 3) << 4) + (laneHi << 3);
      a[i] = (k < 10) ? (_Float16)ec[k] : (_Float16)0.f;
    }
    v8f acc[8];
#pragma unroll
    for (int t = 0; t < 8; ++t) {
      v16h bf = *(const v16h*)&sW1f[(t * 32 + lane) << 4];
      v8f z = {};
      acc[t] = wmma16x16x32(a, bf, z);
    }
    // bias + relu -> hidden tile (f16) in LDS
#pragma unroll
    for (int t = 0; t < 8; ++t) {
#pragma unroll
      for (int v = 0; v < 8; ++v) {
        int m = v + 8 * laneHi, col = t * 16 + ln;
        float hv = fmaxf(acc[t][v] + sb1[col], 0.f);
        hbuf[m * EHS + col] = (_Float16)hv;
      }
    }

    // ---- layer 2: h(16x128) @ We2(128x64), 4 K-steps x 4 N-tiles
    v8f oacc[4];
#pragma unroll
    for (int nt = 0; nt < 4; ++nt) { v8f z = {}; oacc[nt] = z; }
#pragma unroll
    for (int ks = 0; ks < 4; ++ks) {
      const v8h* q = (const v8h*)(hbuf + ln * EHS + ks * 32 + laneHi * 8);
      V16U af;
      af.h[0] = q[0];       // K = ks*32 + laneHi*8 + [0..7]
      af.h[1] = q[2];       // K = ks*32 + laneHi*8 + 16 + [0..7]
#pragma unroll
      for (int nt = 0; nt < 4; ++nt) {
        v16h bf = *(const v16h*)&sW2f[(((ks * 4 + nt) * 32) + lane) << 4];
        oacc[nt] = wmma16x16x32(af.v, bf, oacc[nt]);
      }
    }
    // bias + relu -> unsymmetrized Eh
#pragma unroll
    for (int nt = 0; nt < 4; ++nt) {
#pragma unroll
      for (int v = 0; v < 8; ++v) {
        int m = v + 8 * laneHi;
        int n = nt * 16 + ln;
        float ov = fmaxf(oacc[nt][v] + sb2[n], 0.f);
        EhOut[(((size_t)(b * NN + j)) * NN + (kt * 16 + m)) * EOUT + n] = ov;
      }
    }
  }
}

// ---------------------------------------------------------------- Xh kernel
// 128 threads = 4 waves; wave = one 16-row M-tile of the 8192 node rows.
// sWf is reused: Wx1 fragments for layer 1, then one Wx2 k-slab per step.
#define XHS 264
__global__ __launch_bounds__(128) void xh_kernel(const float* __restrict__ X,
                                                 const float* __restrict__ Wx1,
                                                 const float* __restrict__ bx1,
                                                 const float* __restrict__ Wx2,
                                                 const float* __restrict__ bx2,
                                                 const unsigned char* __restrict__ nmask,
                                                 const int* __restrict__ mp,
                                                 float* __restrict__ XhOut) {
  __shared__ __attribute__((aligned(32))) _Float16 sWf[16 * 32 * 16];   // [tile][lane][i]
  __shared__ __attribute__((aligned(32))) _Float16 sH[4][16 * XHS];
  __shared__ float sb1[HX];
  __shared__ float sb2[DXH];

  int tid = threadIdx.x;
  // stage Wx1 (32x256) in fragment-major layout (16 N-tiles)
  for (int idx = tid; idx < 16 * 32 * 16; idx += 128) {
    int i = idx & 15, lane_s = (idx >> 4) & 31, t = idx >> 9;
    int k = (i & 7) + 16 * (i >> 3) + 8 * (lane_s >> 4);
    int col = t * 16 + (lane_s & 15);
    sWf[idx] = (_Float16)Wx1[k * HX + col];
  }
  for (int idx = tid; idx < HX; idx += 128) { sb1[idx] = bx1[idx]; sb2[idx] = bx2[idx]; }
  __syncthreads();

  int wave = tid >> 5, lane = tid & 31;
  int laneHi = lane >> 4, ln = lane & 15;
  int rt = blockIdx.x * 4 + wave;
  int r0 = rt * 16;
  int b = r0 >> 7, i0 = r0 & 127;
  int irow = i0 + ln;
  int mi = mp[b * NN + irow];

  float xc[32];
  const float* xp = X + ((size_t)(b * NN + irow)) * DXIN;
#pragma unroll
  for (int c = 0; c < 16; ++c) xc[c] = xp[c];
  if (mi >= 0) {
    const float* xp2 = X + ((size_t)(b * NN + mi)) * DXIN;
#pragma unroll
    for (int c = 0; c < 16; ++c) xc[16 + c] = xp2[c];
  } else {
#pragma unroll
    for (int c = 0; c < 16; ++c) xc[16 + c] = 0.f;
  }
  v16h a;
#pragma unroll
  for (int i = 0; i < 16; ++i) {
    int k = (i & 7) + ((i >> 3) << 4) + (laneHi << 3);
    a[i] = (_Float16)xc[k];
  }
  _Float16* hbuf = sH[wave];

  // ---- layer 1: Xc(16x32) @ Wx1(32x256), 16 N-tiles (two halves for regs)
  for (int half = 0; half < 2; ++half) {
    v8f acc[8];
#pragma unroll
    for (int t = 0; t < 8; ++t) {
      int tt = half * 8 + t;
      v16h bf = *(const v16h*)&sWf[(tt * 32 + lane) << 4];
      v8f z = {};
      acc[t] = wmma16x16x32(a, bf, z);
    }
#pragma unroll
    for (int t = 0; t < 8; ++t) {
#pragma unroll
      for (int v = 0; v < 8; ++v) {
        int m = v + 8 * laneHi, col = (half * 8 + t) * 16 + ln;
        float hv = fmaxf(acc[t][v] + sb1[col], 0.f);
        hbuf[m * XHS + col] = (_Float16)hv;
      }
    }
  }

  // ---- layer 2: h(16x256) @ Wx2(256x256); ks-outer, staged k-slabs in sWf
  v8f oacc[16];
#pragma unroll
  for (int nt = 0; nt < 16; ++nt) { v8f z = {}; oacc[nt] = z; }
  for (int ks = 0; ks < 8; ++ks) {
    __syncthreads();   // previous slab (or Wx1 frags) no longer needed
    for (int idx = tid; idx < 16 * 32 * 16; idx += 128) {
      int i = idx & 15, lane_s = (idx >> 4) & 31, nt = idx >> 9;
      int k = ks * 32 + (i & 7) + 16 * (i >> 3) + 8 * (lane_s >> 4);
      int col = nt * 16 + (lane_s & 15);
      sWf[idx] = (_Float16)Wx2[(size_t)k * DXH + col];
    }
    __syncthreads();
    const v8h* q = (const v8h*)(hbuf + ln * XHS + ks * 32 + laneHi * 8);
    V16U af;
    af.h[0] = q[0];
    af.h[1] = q[2];
#pragma unroll
    for (int nt = 0; nt < 16; ++nt) {
      v16h bf = *(const v16h*)&sWf[(nt * 32 + lane) << 4];
      oacc[nt] = wmma16x16x32(af.v, bf, oacc[nt]);
    }
  }
#pragma unroll
  for (int nt = 0; nt < 16; ++nt) {
#pragma unroll
    for (int v = 0; v < 8; ++v) {
      int m = v + 8 * laneHi;
      int n = nt * 16 + ln;
      int gi = b * NN + i0 + m;
      float msk = nmask[gi] ? 1.f : 0.f;
      XhOut[(size_t)gi * DXH + n] = fmaxf(oacc[nt][v] + sb2[n], 0.f) * msk;
    }
  }
}

// ---------------------------------------------------------------- y kernel
__global__ __launch_bounds__(128) void y_kernel(const float* __restrict__ y,
                                                const float* __restrict__ Wy1,
                                                const float* __restrict__ by1,
                                                const float* __restrict__ Wy2,
                                                const float* __restrict__ by2,
                                                float* __restrict__ yhOut) {
  __shared__ float sh[HY];
  int b = blockIdx.x, t = threadIdx.x;
  if (t < HY) {
    float acc = by1[t];
    for (int d = 0; d < 12; ++d) acc += y[b * 12 + d] * Wy1[d * HY + t];
    sh[t] = fmaxf(acc, 0.f);
  }
  __syncthreads();
  if (t < YOUT) {
    float acc = by2[t];
    for (int h = 0; h < HY; ++h) acc += sh[h] * Wy2[h * YOUT + t];
    yhOut[b * YOUT + t] = fmaxf(acc, 0.f);
  }
}

// ---------------------------------------------------------------- symmetrize
// block = (j, b); handles k in [j,128), disjoint in-place RMW across blocks
__global__ __launch_bounds__(256) void sym_kernel(float* __restrict__ Eh,
                                                  const unsigned char* __restrict__ nmask) {
  int j = blockIdx.x, b = blockIdx.y;
  int tid = threadIdx.x;
  int koff = tid >> 6;        // 0..3
  int c = tid & 63;           // channel
  float mj = nmask[b * NN + j] ? 1.f : 0.f;
  for (int k0 = j; k0 < NN; k0 += 4) {
    int k = k0 + koff;
    if (k >= NN) continue;
    float mk = nmask[b * NN + k] ? 1.f : 0.f;
    size_t i1 = (((size_t)(b * NN + j)) * NN + k) * EOUT + c;
    size_t i2 = (((size_t)(b * NN + k)) * NN + j) * EOUT + c;
    if (k == j) {
      Eh[i1] = Eh[i1] * mj * mk;
    } else {
      float v = 0.5f * (Eh[i1] + Eh[i2]) * mj * mk;
      Eh[i1] = v;
      Eh[i2] = v;
    }
  }
}

// ---------------------------------------------------------------- launcher
extern "C" void kernel_launch(void* const* d_in, const int* in_sizes, int n_in,
                              void* d_out, int out_size, void* d_ws, size_t ws_size,
                              hipStream_t stream) {
  (void)in_sizes; (void)n_in; (void)out_size; (void)ws_size;
  const float* X   = (const float*)d_in[0];
  const float* E   = (const float*)d_in[1];
  const float* y   = (const float*)d_in[2];
  const unsigned char* nmask = (const unsigned char*)d_in[6];
  const int*   am  = (const int*)d_in[7];
  const int*   mol = (const int*)d_in[9];
  const float* Wx1 = (const float*)d_in[10];
  const float* bx1 = (const float*)d_in[11];
  const float* Wx2 = (const float*)d_in[12];
  const float* bx2 = (const float*)d_in[13];
  const float* We1 = (const float*)d_in[14];
  const float* be1 = (const float*)d_in[15];
  const float* We2 = (const float*)d_in[16];
  const float* be2 = (const float*)d_in[17];
  const float* Wy1 = (const float*)d_in[18];
  const float* by1 = (const float*)d_in[19];
  const float* Wy2 = (const float*)d_in[20];
  const float* by2 = (const float*)d_in[21];

  float* Xh = (float*)d_out;                              // 64*128*256
  float* Eh = Xh + (size_t)NB * NN * DXH;                 // 64*128*128*64
  float* yh = Eh + (size_t)NB * NN * NN * EOUT;           // 64*64
  int* mp = (int*)d_ws;                                   // 64*128 ints

  map_kernel<<<dim3(NB), dim3(NN), 0, stream>>>(am, mol, mp);
  eh_kernel<<<dim3(NN / 16, NN / 16, NB), dim3(256), 0, stream>>>(E, We1, be1, We2, be2, mp, Eh);
  xh_kernel<<<dim3((NB * NN) / 64), dim3(128), 0, stream>>>(X, Wx1, bx1, Wx2, bx2, nmask, mp, Xh);
  y_kernel<<<dim3(NB), dim3(128), 0, stream>>>(y, Wy1, by1, Wy2, by2, yh);
  sym_kernel<<<dim3(NN, NB), dim3(256), 0, stream>>>(Eh, nmask);
}
